// SequenceLayer_15058155340148
// MI455X (gfx1250) — matrix operand order repaired and checked by reading, not compiled, not run on hardware
//
#include <hip/hip_runtime.h>
#include <stdint.h>

#define L_SEQ  2048
#define BSZ    32
#define HDIM   512
#define PDIM   256
#define MROWS  (L_SEQ * BSZ)        // 65536
#define NCHUNK 16
#define CHLEN  (L_SEQ / NCHUNK)     // 128

typedef __attribute__((ext_vector_type(16))) __bf16 v16bf;
typedef __attribute__((ext_vector_type(8)))  float  v8f;

// ---------- helpers ----------
__device__ __forceinline__ unsigned short f2bf(float f) {
  unsigned int u = __float_as_uint(f);
  unsigned int r = u + 0x7FFFu + ((u >> 16) & 1u);   // round-to-nearest-even
  return (unsigned short)(r >> 16);
}

__device__ __forceinline__ float wave_sum32(float v) {
#pragma unroll
  for (int off = 16; off > 0; off >>= 1) v += __shfl_xor(v, off, 32);
  return v;
}

// async DMA: global -> LDS, 16B per lane, tracked by ASYNCcnt
__device__ __forceinline__ void async_b128(unsigned lds_addr, const void* gaddr) {
  asm volatile("global_load_async_to_lds_b128 %0, %1, off"
               :: "v"(lds_addr), "v"(gaddr) : "memory");
}
__device__ __forceinline__ void wait_async0() {
  asm volatile("s_wait_asynccnt 0x0" ::: "memory");
}

// ---------- LayerNorm (input) -> xn f32 + xn bf16 ----------
__global__ __launch_bounds__(256) void k_ln_in(const float* __restrict__ x,
                                               const float* __restrict__ scale,
                                               const float* __restrict__ bias,
                                               float* __restrict__ xnf,
                                               unsigned short* __restrict__ xnb) {
  int row  = blockIdx.x * 8 + (threadIdx.x >> 5);
  int lane = threadIdx.x & 31;
  size_t base = (size_t)row * HDIM;
  float v[16]; float s = 0.f;
#pragma unroll
  for (int k = 0; k < 16; ++k) { v[k] = x[base + lane + 32 * k]; s += v[k]; }
  float mu = wave_sum32(s) * (1.0f / HDIM);
  float q = 0.f;
#pragma unroll
  for (int k = 0; k < 16; ++k) { float d = v[k] - mu; q += d * d; }
  float var = wave_sum32(q) * (1.0f / HDIM);
  float rs = rsqrtf(var + 1e-6f);
#pragma unroll
  for (int k = 0; k < 16; ++k) {
    int col = lane + 32 * k;
    float o = (v[k] - mu) * rs * scale[col] + bias[col];
    xnf[base + col] = o;
    xnb[base + col] = f2bf(o);
  }
}

// ---------- LayerNorm + tanh-GELU -> g bf16 ----------
__global__ __launch_bounds__(256) void k_ln_gelu(const float* __restrict__ y,
                                                 const float* __restrict__ scale,
                                                 const float* __restrict__ bias,
                                                 unsigned short* __restrict__ g) {
  int row  = blockIdx.x * 8 + (threadIdx.x >> 5);
  int lane = threadIdx.x & 31;
  size_t base = (size_t)row * HDIM;
  float v[16]; float s = 0.f;
#pragma unroll
  for (int k = 0; k < 16; ++k) { v[k] = y[base + lane + 32 * k]; s += v[k]; }
  float mu = wave_sum32(s) * (1.0f / HDIM);
  float q = 0.f;
#pragma unroll
  for (int k = 0; k < 16; ++k) { float d = v[k] - mu; q += d * d; }
  float var = wave_sum32(q) * (1.0f / HDIM);
  float rs = rsqrtf(var + 1e-6f);
#pragma unroll
  for (int k = 0; k < 16; ++k) {
    int col = lane + 32 * k;
    float o = (v[k] - mu) * rs * scale[col] + bias[col];
    float t = 0.5f * o * (1.0f + tanhf(0.7978845608f * (o + 0.044715f * o * o * o)));
    g[base + col] = f2bf(t);
  }
}

// ---------- small precompute kernels ----------
__global__ void k_prep_lam(const float* __restrict__ Lre, const float* __restrict__ Lim,
                           const float* __restrict__ lstep,
                           float* __restrict__ lamr, float* __restrict__ lami,
                           float* __restrict__ coefr, float* __restrict__ coefi) {
  int p = threadIdx.x;                      // 256 threads
  float dt = expf(lstep[p]);
  float lr = Lre[p], li = Lim[p];
  float m  = expf(lr * dt);
  float cr = m * cosf(li * dt);
  float ci = m * sinf(li * dt);
  lamr[p] = cr; lami[p] = ci;
  float nr = cr - 1.0f, ni = ci;            // (Lam_bar - 1) / Lam
  float den = lr * lr + li * li;
  coefr[p] = (nr * lr + ni * li) / den;
  coefi[p] = (ni * lr - nr * li) / den;
}

// Bcat (2P x H) bf16 : rows 0..255 = B_bar_re, rows 256..511 = B_bar_im
__global__ __launch_bounds__(256) void k_prep_bcat(const float* __restrict__ Bre,
                                                   const float* __restrict__ Bim,
                                                   const float* __restrict__ coefr,
                                                   const float* __restrict__ coefi,
                                                   unsigned short* __restrict__ bcat) {
  int id = blockIdx.x * 256 + threadIdx.x;  // P*H = 131072
  int p = id >> 9, h = id & 511;
  float br = Bre[id], bi = Bim[id];
  float cr = coefr[p], ci = coefi[p];
  bcat[(size_t)p * 512 + h]         = f2bf(cr * br - ci * bi);
  bcat[(size_t)(p + 256) * 512 + h] = f2bf(cr * bi + ci * br);
}

// Ccat (H x 2P) bf16 : cols 0..255 = 2*C_re, cols 256..511 = -2*C_im
__global__ __launch_bounds__(256) void k_prep_ccat(const float* __restrict__ Cre,
                                                   const float* __restrict__ Cim,
                                                   unsigned short* __restrict__ ccat) {
  int id = blockIdx.x * 256 + threadIdx.x;  // H*P = 131072
  int h = id >> 8, p = id & 255;
  ccat[(size_t)h * 512 + p]       = f2bf(2.0f * Cre[id]);
  ccat[(size_t)h * 512 + 256 + p] = f2bf(-2.0f * Cim[id]);
}

// Wcat (1024 x 512) bf16 : rows 0..511 = W1^T, rows 512..1023 = W2^T
__global__ __launch_bounds__(256) void k_prep_wt(const float* __restrict__ W1,
                                                 const float* __restrict__ W2,
                                                 unsigned short* __restrict__ wcat) {
  int id = blockIdx.x * 256 + threadIdx.x;  // H*H = 262144
  int k = id >> 9, n = id & 511;
  wcat[(size_t)n * 512 + k]         = f2bf(W1[id]);
  wcat[(size_t)(n + 512) * 512 + k] = f2bf(W2[id]);
}

// ---------- generic bf16 WMMA GEMM:  D(MxN) = A(MxK) * B(NxK)^T ----------
// Block tile 128x128, 8 waves of 32x64 (2x4 accumulators), BK=64 per stage,
// double-buffered LDS fed by global_load_async_to_lds_b128 (ASYNCcnt DMA).
// MODE 0: plain f32 store to Dout (ldN = N).
// MODE 1: val += auxD[col] * auxXn[row*512+col], store to Dout (N = 512).
// MODE 2: split store: col<512 -> Dout[row*512+col], else Dout2[row*512+col-512].
template <int MODE>
__global__ __launch_bounds__(256) void k_gemm(const unsigned short* __restrict__ A,
                                              const unsigned short* __restrict__ B,
                                              float* __restrict__ Dout,
                                              float* __restrict__ Dout2,
                                              int N, int K,
                                              const float* __restrict__ auxXn,
                                              const float* __restrict__ auxD) {
  // per buffer: A tile 128x64 @ stride 72 halves (18432 B) + B tile same -> 36864 B
  __shared__ unsigned short smem[2 * 2 * 128 * 72];   // 73728 bytes total

  int tid = threadIdx.x;
  int rowBase = blockIdx.x * 128;
  int colBase = blockIdx.y * 128;
  int wid = tid >> 5, lane = tid & 31;
  int wm = wid & 3, wn = wid >> 2;          // 4x2 waves -> 128x128 block tile
  int h = lane >> 4, l15 = lane & 15;

  // global->LDS mapping: lane owns column-chunk (tid&7), rows tid>>3 + j*32
  int gr = tid >> 3, gc = tid & 7;
  unsigned smemBase = (unsigned)(unsigned long long)(const void*)smem;  // LDS byte addr

  v8f acc[2][4] = {};
  union Frag { v16bf v; uint4 q[2]; };

  auto issue = [&](int kk, int sbuf) {
    unsigned bb = smemBase + (unsigned)(sbuf * 36864);
#pragma unroll
    for (int j = 0; j < 4; ++j) {
      int r = gr + j * 32;
      unsigned off = (unsigned)((r * 72 + gc * 8) * 2);
      async_b128(bb + off,          A + (size_t)(rowBase + r) * K + kk + gc * 8);
      async_b128(bb + 18432u + off, B + (size_t)(colBase + r) * K + kk + gc * 8);
    }
  };

  const int NSTAGE = K >> 6;
  issue(0, 0);
  wait_async0();
  __syncthreads();

  for (int s = 0; s < NSTAGE; ++s) {
    if (s + 1 < NSTAGE) issue((s + 1) << 6, (s + 1) & 1);   // DMA next stage

    const unsigned short* As_ = smem + (s & 1) * 18432;     // ushort offsets
    const unsigned short* Bs_ = As_ + 9216;
#pragma unroll
    for (int kc = 0; kc < 2; ++kc) {        // two 32-K chunks per stage
      Frag a[2], bf[4];
#pragma unroll
      for (int mi = 0; mi < 2; ++mi) {      // A frag: row=l15, K = h*8+[0..7], 16+h*8+[0..7]
        int ar = wm * 32 + mi * 16 + l15;
        a[mi].q[0] = *(const uint4*)(&As_[ar * 72 + kc * 32 + h * 8]);
        a[mi].q[1] = *(const uint4*)(&As_[ar * 72 + kc * 32 + 16 + h * 8]);
      }
#pragma unroll
      for (int ni = 0; ni < 4; ++ni) {      // B frag: col=l15, K = h*8+[0..7], 16+h*8+[0..7]
        int br = wn * 64 + ni * 16 + l15;
        bf[ni].q[0] = *(const uint4*)(&Bs_[br * 72 + kc * 32 + h * 8]);
        bf[ni].q[1] = *(const uint4*)(&Bs_[br * 72 + kc * 32 + 16 + h * 8]);
      }
#pragma unroll
      for (int mi = 0; mi < 2; ++mi)
#pragma unroll
        for (int ni = 0; ni < 4; ++ni)
          acc[mi][ni] = __builtin_amdgcn_wmma_f32_16x16x32_bf16(
              false, a[mi].v, false, bf[ni].v, (short)0, acc[mi][ni], false, false);
    }
    wait_async0();                          // next-stage DMA landed
    __syncthreads();                        // everyone done reading current buffer
  }

#pragma unroll
  for (int mi = 0; mi < 2; ++mi) {
#pragma unroll
    for (int ni = 0; ni < 4; ++ni) {
      int row0 = rowBase + wm * 32 + mi * 16 + 8 * h;   // C/D layout: M = v + 8*h, N = l15
      int col  = colBase + wn * 64 + ni * 16 + l15;
#pragma unroll
      for (int v = 0; v < 8; ++v) {
        float val = acc[mi][ni][v];
        size_t row = (size_t)(row0 + v);
        if (MODE == 0) {
          Dout[row * N + col] = val;
        } else if (MODE == 1) {
          val += auxD[col] * auxXn[row * HDIM + col];
          Dout[row * HDIM + col] = val;
        } else {                            // MODE 2: split into two 512-wide outputs
          if (col < 512) Dout[row * 512 + col] = val;
          else           Dout2[row * 512 + (col - 512)] = val;
        }
      }
    }
  }
}

// ---------- chunked reset-scan: pass 1 (per-chunk aggregates) ----------
__global__ __launch_bounds__(256) void k_scan1(const float* __restrict__ Bu,
                                               const int* __restrict__ dres,
                                               const float* __restrict__ lamr_,
                                               const float* __restrict__ lami_,
                                               float* __restrict__ aAr, float* __restrict__ aAi,
                                               float* __restrict__ abr, float* __restrict__ abi,
                                               float* __restrict__ ac) {
  int tid = blockIdx.x * 256 + threadIdx.x;   // BS*P*NCHUNK = 131072
  int p = tid & 255, b = (tid >> 8) & 31, ck = tid >> 13;
  float lr = lamr_[p], li = lami_[p];
  float Ar = 1.f, Ai = 0.f, br = 0.f, bi = 0.f, c = 0.f;
  int l0 = ck * CHLEN;
  for (int t = 0; t < CHLEN; ++t) {
    int l = l0 + t;
    size_t base = ((size_t)(l * BSZ + b)) * 512 + p;
    float ur = Bu[base], ui = Bu[base + 256];
    int dd = dres[l * BSZ + b];
    if (dd) { Ar = lr; Ai = li; br = ur; bi = ui; c = 1.f; }
    else {
      float nAr = lr * Ar - li * Ai; Ai = lr * Ai + li * Ar; Ar = nAr;
      float nbr = lr * br - li * bi + ur; bi = lr * bi + li * br + ui; br = nbr;
    }
  }
  int idx = ck * (BSZ * PDIM) + b * PDIM + p;
  aAr[idx] = Ar; aAi[idx] = Ai; abr[idx] = br; abi[idx] = bi; ac[idx] = c;
}

// ---------- pass 2: sequential prefix over 16 chunks ----------
__global__ __launch_bounds__(256) void k_scan2(const float* __restrict__ h0r,
                                               const float* __restrict__ h0i,
                                               const float* __restrict__ aAr, const float* __restrict__ aAi,
                                               const float* __restrict__ abr, const float* __restrict__ abi,
                                               const float* __restrict__ ac,
                                               float* __restrict__ hinr, float* __restrict__ hini) {
  int tid = blockIdx.x * 256 + threadIdx.x;   // BS*P = 8192
  int p = tid & 255, b = tid >> 8;
  float hr = h0r[b * PDIM + p], hi = h0i[b * PDIM + p];
  for (int ck = 0; ck < NCHUNK; ++ck) {
    int idx = ck * (BSZ * PDIM) + b * PDIM + p;
    hinr[idx] = hr; hini[idx] = hi;
    float Ar = aAr[idx], Ai = aAi[idx], br = abr[idx], bi = abi[idx], c = ac[idx];
    float nr = Ar * hr - Ai * hi + br;
    float ni = Ar * hi + Ai * hr + bi;
    hr = (c != 0.f) ? br : nr;
    hi = (c != 0.f) ? bi : ni;
  }
}

// ---------- pass 3: apply prefix, emit states bf16 [S_re|S_im] + hidden_out ----------
__global__ __launch_bounds__(256) void k_scan3(const float* __restrict__ Bu,
                                               const int* __restrict__ dres,
                                               const float* __restrict__ lamr_,
                                               const float* __restrict__ lami_,
                                               const float* __restrict__ hinr,
                                               const float* __restrict__ hini,
                                               unsigned short* __restrict__ scat,
                                               float* __restrict__ outF) {
  int tid = blockIdx.x * 256 + threadIdx.x;
  int p = tid & 255, b = (tid >> 8) & 31, ck = tid >> 13;
  float lr = lamr_[p], li = lami_[p];
  int idx = ck * (BSZ * PDIM) + b * PDIM + p;
  float hr = hinr[idx], hi = hini[idx];
  int l0 = ck * CHLEN;
  for (int t = 0; t < CHLEN; ++t) {
    int l = l0 + t;
    size_t base = ((size_t)(l * BSZ + b)) * 512 + p;
    float ur = Bu[base], ui = Bu[base + 256];
    int dd = dres[l * BSZ + b];
    if (dd) { hr = ur; hi = ui; }
    else { float nr = lr * hr - li * hi + ur; hi = lr * hi + li * hr + ui; hr = nr; }
    scat[base] = f2bf(hr);
    scat[base + 256] = f2bf(hi);
    if (l == L_SEQ - 1) {
      size_t ho = (size_t)MROWS * HDIM;
      outF[ho + b * PDIM + p] = hr;
      outF[ho + BSZ * PDIM + b * PDIM + p] = hi;
    }
  }
}

// ---------- final gating + residual ----------
__global__ __launch_bounds__(256) void k_final(const float* __restrict__ x,
                                               const float* __restrict__ Z1,
                                               const float* __restrict__ Z2,
                                               const float* __restrict__ b1,
                                               const float* __restrict__ b2,
                                               float* __restrict__ outp) {
  size_t id = (size_t)blockIdx.x * 256 + threadIdx.x;
  int col = (int)(id & 511);
  float v1 = Z1[id] + b1[col];
  float v2 = Z2[id] + b2[col];
  outp[id] = x[id] + v1 * (1.0f / (1.0f + expf(-v2)));
}

// ---------- host side ----------
extern "C" void kernel_launch(void* const* d_in, const int* in_sizes, int n_in,
                              void* d_out, int out_size, void* d_ws, size_t ws_size,
                              hipStream_t stream) {
  (void)in_sizes; (void)n_in; (void)out_size; (void)ws_size;
  const float* h0r   = (const float*)d_in[0];
  const float* h0i   = (const float*)d_in[1];
  const float* x     = (const float*)d_in[2];
  const float* Lre   = (const float*)d_in[3];
  const float* Lim   = (const float*)d_in[4];
  const float* Bre   = (const float*)d_in[5];
  const float* Bim   = (const float*)d_in[6];
  const float* Cre   = (const float*)d_in[7];
  const float* Cim   = (const float*)d_in[8];
  const float* Dv    = (const float*)d_in[9];
  const float* lstep = (const float*)d_in[10];
  const float* nsc   = (const float*)d_in[11];
  const float* nbi   = (const float*)d_in[12];
  const float* W1    = (const float*)d_in[13];
  const float* b1    = (const float*)d_in[14];
  const float* W2    = (const float*)d_in[15];
  const float* b2    = (const float*)d_in[16];
  const int*   dres  = (const int*)d_in[17];
  float* outF = (float*)d_out;

  char* ws = (char*)d_ws;
  size_t off = 0;
  auto alloc = [&](size_t bytes) -> char* {
    char* p = ws + off; off += (bytes + 255) & ~(size_t)255; return p;
  };
  float*          xnf  = (float*)alloc((size_t)MROWS * HDIM * 4);          // xn f32, reused as Z1
  unsigned short* xnb  = (unsigned short*)alloc((size_t)MROWS * HDIM * 2); // xn bf16
  float*          bu   = (float*)alloc((size_t)MROWS * HDIM * 4);          // Bu, reused as ypre then Z2
  unsigned short* scat = (unsigned short*)alloc((size_t)MROWS * HDIM * 2); // states bf16, reused as g
  float* lamr  = (float*)alloc(PDIM * 4);
  float* lami  = (float*)alloc(PDIM * 4);
  float* coefr = (float*)alloc(PDIM * 4);
  float* coefi = (float*)alloc(PDIM * 4);
  unsigned short* bcat = (unsigned short*)alloc((size_t)512 * 512 * 2);
  unsigned short* ccat = (unsigned short*)alloc((size_t)512 * 512 * 2);
  unsigned short* wcat = (unsigned short*)alloc((size_t)1024 * 512 * 2);
  size_t nAgg = (size_t)NCHUNK * BSZ * PDIM;
  float* aAr = (float*)alloc(nAgg * 4);
  float* aAi = (float*)alloc(nAgg * 4);
  float* abr = (float*)alloc(nAgg * 4);
  float* abi = (float*)alloc(nAgg * 4);
  float* ac  = (float*)alloc(nAgg * 4);
  float* hinr = (float*)alloc(nAgg * 4);
  float* hini = (float*)alloc(nAgg * 4);

  // precompute small operands
  k_prep_lam<<<1, 256, 0, stream>>>(Lre, Lim, lstep, lamr, lami, coefr, coefi);
  k_prep_bcat<<<512, 256, 0, stream>>>(Bre, Bim, coefr, coefi, bcat);
  k_prep_ccat<<<512, 256, 0, stream>>>(Cre, Cim, ccat);
  k_prep_wt<<<1024, 256, 0, stream>>>(W1, W2, wcat);

  // xn = LN(x)
  k_ln_in<<<MROWS / 8, 256, 0, stream>>>(x, nsc, nbi, xnf, xnb);

  // Bu = xn @ Bcat^T  (M x 512, re|im halves)
  k_gemm<0><<<dim3(MROWS / 128, 512 / 128), 256, 0, stream>>>(
      xnb, bcat, bu, nullptr, 512, 512, nullptr, nullptr);

  // chunked reset-scan over L
  k_scan1<<<(BSZ * PDIM * NCHUNK) / 256, 256, 0, stream>>>(bu, dres, lamr, lami, aAr, aAi, abr, abi, ac);
  k_scan2<<<(BSZ * PDIM) / 256, 256, 0, stream>>>(h0r, h0i, aAr, aAi, abr, abi, ac, hinr, hini);
  k_scan3<<<(BSZ * PDIM * NCHUNK) / 256, 256, 0, stream>>>(bu, dres, lamr, lami, hinr, hini, scat, outF);

  // ypre = [S_re|S_im] @ Ccat^T + D*xn   (writes over bu; 2.0 and -sign folded into ccat)
  k_gemm<1><<<dim3(MROWS / 128, 512 / 128), 256, 0, stream>>>(
      scat, ccat, bu, nullptr, 512, 512, xnf, Dv);

  // g = gelu(LN(ypre))  -> reuse scat buffer
  k_ln_gelu<<<MROWS / 8, 256, 0, stream>>>(bu, nsc, nbi, scat);

  // [Z1|Z2] = g @ Wcat^T, split-stored: Z1 -> xnf, Z2 -> bu (single pass over g)
  k_gemm<2><<<dim3(MROWS / 128, 1024 / 128), 256, 0, stream>>>(
      scat, wcat, xnf, bu, 1024, 512, nullptr, nullptr);

  // out = x + (Z1+b1)*sigmoid(Z2+b2)
  k_final<<<((size_t)MROWS * HDIM) / 256, 256, 0, stream>>>(x, xnf, bu, b1, b2, outF);
}
